// GraphResAttentionLayer_3410204033348
// MI455X (gfx1250) — matrix operand order repaired and checked by reading, not compiled, not run on hardware
//
#include <hip/hip_runtime.h>
#include <hip/hip_bf16.h>

// GAT layer for MI455X (gfx1250, wave32).
// N=8192 nodes, IN=OUT=128.
// Roofline: adj (256MB) read exactly once (k_pack, b128 streams) -> ~11us at
// 23.3 TB/s. h (4MB), hT (2MB), vbits (8MB) are L2-resident; s_dst in LDS.
// att@h (17.2 GFLOP) via v_wmma_f32_16x16x32_f16; X@W via v_wmma_f32_16x16x4_f32.
// k_attn: 4 waves cooperate per 16-row block (2048 waves total) -- j-range is
// split 4-way, softmax stats combined via LDS, partial WMMA accumulators
// merged with ds_add_f32.

#define NN    8192
#define DIM   128
#define THR   0.0001f
#define ALPHA 0.2f
#define NEGV  -1000000000.0f
#define SPLIT 4

typedef __attribute__((ext_vector_type(2)))  float    v2f;
typedef __attribute__((ext_vector_type(8)))  float    v8f;
typedef __attribute__((ext_vector_type(16))) _Float16 v16h;

// ---------------------------------------------------------------------------
// K1: h = input @ W   (f32 WMMA 16x16x4, K=128 -> 32 wmma per 16x16 tile)
// ---------------------------------------------------------------------------
__global__ void __launch_bounds__(128) k_gemm_h(const float* __restrict__ inp,
                                                const float* __restrict__ W,
                                                float* __restrict__ h) {
    const int wave = threadIdx.x >> 5;
    const int lane = threadIdx.x & 31;
    const int tile = blockIdx.x * 4 + wave;       // 512*8 = 4096 tiles
    const int m0 = (tile >> 3) * 16;
    const int n0 = (tile & 7) * 16;
    const int half = lane >> 4;
    const int l16  = lane & 15;

    v8f c = {};
    for (int k0 = 0; k0 < DIM; k0 += 4) {
        v2f a, b;
        const float* ap = inp + (m0 + l16) * DIM + k0 + 2 * half;
        a.x = ap[0];
        a.y = ap[1];
        const float* bp = W + (k0 + 2 * half) * DIM + n0 + l16;
        b.x = bp[0];
        b.y = bp[DIM];
        c = __builtin_amdgcn_wmma_f32_16x16x4_f32(false, a, false, b,
                                                  (short)0, c, false, false);
    }
    const int col   = n0 + l16;
    const int rbase = m0 + 8 * half;
#pragma unroll
    for (int v = 0; v < 8; v++)
        h[(rbase + v) * DIM + col] = c[v];
}

// ---------------------------------------------------------------------------
// K2: per-row stats: s_src = h@a[:128], s_dst = h@a[128:], rowsum = sum(h),
//     plus f16 transpose hT[n][j] = (f16)h[j][n]. One wave per row.
// ---------------------------------------------------------------------------
__global__ void __launch_bounds__(256) k_rowstats(const float* __restrict__ h,
                                                  const float* __restrict__ a,
                                                  float* __restrict__ s_src,
                                                  float* __restrict__ s_dst,
                                                  float* __restrict__ rowsum,
                                                  _Float16* __restrict__ hT) {
    const int lane = threadIdx.x & 31;
    const int row  = blockIdx.x * 8 + (threadIdx.x >> 5);

    float sum = 0.f, dsrc = 0.f, ddst = 0.f;
#pragma unroll
    for (int c = lane; c < DIM; c += 32) {
        float v = h[row * DIM + c];
        sum  += v;
        dsrc += v * a[c];
        ddst += v * a[DIM + c];
        hT[(size_t)c * NN + row] = (_Float16)v;
    }
#pragma unroll
    for (int off = 16; off > 0; off >>= 1) {
        sum  += __shfl_xor(sum,  off, 32);
        dsrc += __shfl_xor(dsrc, off, 32);
        ddst += __shfl_xor(ddst, off, 32);
    }
    if (lane == 0) {
        s_src[row]  = dsrc;
        s_dst[row]  = ddst;
        rowsum[row] = sum;
    }
}

// ---------------------------------------------------------------------------
// K3: pack validity bitmask -- THE roofline kernel (256MB of adj, read once).
// b128 loads: each lane handles 4 consecutive j (int4+float4), builds a
// validity nibble; 3 shuffle-OR steps assemble each 32-bit word from 8 lanes.
// ---------------------------------------------------------------------------
__global__ void __launch_bounds__(256) k_pack(const int* __restrict__ adj,
                                              const float* __restrict__ rowsum,
                                              unsigned* __restrict__ vbits) {
    const int lane = threadIdx.x & 31;
    const int row  = blockIdx.x * 8 + (threadIdx.x >> 5);
    const int4*   arow = (const int4*)(adj + (size_t)row * NN);
    const float4* rs   = (const float4*)rowsum;

    for (int it = 0; it < NN / 128; it++) {       // 64 iterations, 512B/wave
        int idx = it * 32 + lane;                 // int4 index
        int4   av = arow[idx];
        float4 rv = rs[idx];
        unsigned nib = ((av.x > 0 && rv.x != 0.f) ? 1u : 0u)
                     | ((av.y > 0 && rv.y != 0.f) ? 2u : 0u)
                     | ((av.z > 0 && rv.z != 0.f) ? 4u : 0u)
                     | ((av.w > 0 && rv.w != 0.f) ? 8u : 0u);
        unsigned val = nib << (4 * (lane & 7));
        val |= __shfl_xor(val, 1, 32);
        val |= __shfl_xor(val, 2, 32);
        val |= __shfl_xor(val, 4, 32);
        if ((lane & 7) == 0)
            vbits[row * (NN / 32) + it * 4 + (lane >> 3)] = val;
    }
}

// ---------------------------------------------------------------------------
// K4: attention + h' + residual ELU.  One 16-row block per workgroup,
// SPLIT=4 waves each owning a 2048-wide j-quarter (2048 waves chip-wide).
// Lane l handles row l&15 (WMMA A layout); lane pair (l,l+16) splits the
// wave's j-range; cross-wave softmax stats via LDS; partial WMMA
// accumulators merged with ds_add_f32 into a shared 16x128 tile.
// ---------------------------------------------------------------------------
__global__ void __launch_bounds__(128) k_attn(const float* __restrict__ h,
                                              const _Float16* __restrict__ hT,
                                              const float* __restrict__ s_src_g,
                                              const float* __restrict__ s_dst_g,
                                              const unsigned* __restrict__ vbits,
                                              float* __restrict__ out) {
    __shared__ float sdst[NN];            // 32 KB
    __shared__ float lacc[16 * DIM];      // 8 KB partial-accumulator merge
    __shared__ float red_m[SPLIT * 16];
    __shared__ float red_s[SPLIT * 16];
    __shared__ int   red_L[SPLIT * 16];

    for (int t = threadIdx.x; t < NN; t += 128) sdst[t] = s_dst_g[t];
    for (int t = threadIdx.x; t < 16 * DIM; t += 128) lacc[t] = 0.0f;
    __syncthreads();

    const int wave = threadIdx.x >> 5;
    const int lane = threadIdx.x & 31;
    const int half = lane >> 4;
    const int r    = lane & 15;
    const int i0   = blockIdx.x * 16;
    const int row  = i0 + r;

    const float ssrc = s_src_g[row];
    const unsigned* vb = vbits + (size_t)row * (NN / 32);

    // this wave's word range: 64 words (2048 j); each half-lane scans 32 words
    const int w0 = wave * 64 + half * 32;

    // ---- pass 1a: partial row max over valid entries ----
    float m = NEGV;
    for (int w = w0; w < w0 + 32; w++) {
        unsigned bits = vb[w];
        int jbase = w * 32;
        while (bits) {
            int b = __builtin_ctz(bits);
            bits &= bits - 1;
            float e = ssrc + sdst[jbase + b];
            e = e > 0.f ? e : ALPHA * e;
            m = fmaxf(m, e);
        }
    }
    m = fmaxf(m, __shfl_xor(m, 16, 32));
    if (lane < 16) red_m[wave * 16 + r] = m;
    __syncthreads();
    m = NEGV;
#pragma unroll
    for (int q = 0; q < SPLIT; q++) m = fmaxf(m, red_m[q * 16 + r]);

    // ---- pass 1b: partial sum of exp + valid count (with global max) ----
    float s = 0.f;
    int   L = 0;
    for (int w = w0; w < w0 + 32; w++) {
        unsigned bits = vb[w];
        L += __builtin_popcount(bits);
        int jbase = w * 32;
        while (bits) {
            int b = __builtin_ctz(bits);
            bits &= bits - 1;
            float e = ssrc + sdst[jbase + b];
            e = e > 0.f ? e : ALPHA * e;
            s += __expf(e - m);
        }
    }
    s += __shfl_xor(s, 16, 32);
    L += __shfl_xor(L, 16, 32);
    if (lane < 16) { red_s[wave * 16 + r] = s; red_L[wave * 16 + r] = L; }
    __syncthreads();
    s = 0.f; L = 0;
#pragma unroll
    for (int q = 0; q < SPLIT; q++) { s += red_s[q * 16 + r]; L += red_L[q * 16 + r]; }

    const float inv  = (L > 0) ? 1.0f / s : 0.0f; // top = inv when L>0
    const bool  gate = (L > 0) && (inv > THR);
    const float Lf   = (float)L;

    // ---- pass 2: partial h' = attention @ h via f16 WMMA over j-quarter ----
    v8f acc[8];
#pragma unroll
    for (int c = 0; c < 8; c++) acc[c] = (v8f){};

    const int kbase = 16 * half; // A 16x32 f16 K map: contiguous k = 16*half + p
    const int jlo = wave * (NN / SPLIT);
    const int jhi = jlo + (NN / SPLIT);

    for (int j0 = jlo; j0 < jhi; j0 += 32) {
        unsigned bits = vb[j0 >> 5];

        v16h afrag;
#pragma unroll
        for (int p = 0; p < 16; p++) {
            int K = kbase + p;
            float aval = 0.0f;
            if ((bits >> K) & 1u) {
                float e = ssrc + sdst[j0 + K];
                e = e > 0.f ? e : ALPHA * e;
                float pr = __expf(e - m) * inv;           // att value
                aval = (gate && pr >= THR) ? pr * Lf : 0.0f;
            }
            afrag[p] = (_Float16)aval;
        }

        // B 32x16 f16: lanes 0-15 carry K=0..15, lanes 16-31 K=16..31,
        // column = lane&15 -> 32 contiguous bytes of hT per lane.
#pragma unroll
        for (int c = 0; c < 8; c++) {
            int col = c * 16 + r;
            const _Float16* bp = hT + (size_t)col * NN + j0 + 16 * half;
            v16h bfrag = *(const v16h*)bp;
            acc[c] = __builtin_amdgcn_wmma_f32_16x16x32_f16(
                false, afrag, false, bfrag, (short)0, acc[c], false, false);
        }
    }

    // ---- merge the 4 partial accumulators in LDS (ds_add_f32) ----
#pragma unroll
    for (int c = 0; c < 8; c++) {
        int col = c * 16 + r;
#pragma unroll
        for (int v = 0; v < 8; v++) {
            atomicAdd(&lacc[(v + 8 * half) * DIM + col], acc[c][v]);
        }
    }
    __syncthreads();

    // ---- epilogue: out = elu(h + h'), coalesced across the block ----
    for (int e = threadIdx.x; e < 16 * DIM; e += 128) {
        int rr  = i0 + (e >> 7);
        int col = e & 127;
        float x = h[rr * DIM + col] + lacc[e];
        out[rr * DIM + col] = x > 0.f ? x : (__expf(x) - 1.0f);
    }
}

// ---------------------------------------------------------------------------
extern "C" void kernel_launch(void* const* d_in, const int* in_sizes, int n_in,
                              void* d_out, int out_size, void* d_ws, size_t ws_size,
                              hipStream_t stream) {
    const float* inp = (const float*)d_in[0];
    const float* W   = (const float*)d_in[1];
    const float* a   = (const float*)d_in[2];
    const int*   adj = (const int*)d_in[3];
    float*       out = (float*)d_out;

    char* ws = (char*)d_ws;
    float*     h      = (float*)ws;                                  // 4 MB
    _Float16*  hT     = (_Float16*)(ws + (4u << 20));                // 2 MB
    float*     s_src  = (float*)(ws + (6u << 20));                   // 32 KB
    float*     s_dst  = (float*)(ws + (6u << 20) + (32u << 10));     // 32 KB
    float*     rowsum = (float*)(ws + (6u << 20) + (64u << 10));     // 32 KB
    unsigned*  vbits  = (unsigned*)(ws + (7u << 20));                // 8 MB

    k_gemm_h <<<1024, 128, 0, stream>>>(inp, W, h);
    k_rowstats<<<1024, 256, 0, stream>>>(h, a, s_src, s_dst, rowsum, hT);
    k_pack   <<<1024, 256, 0, stream>>>(adj, rowsum, vbits);
    k_attn   <<< 512, 128, 0, stream>>>(h, hT, s_src, s_dst, vbits, out);
}
